// CRF_7876970020982
// MI455X (gfx1250) — matrix operand order repaired and checked by reading, not compiled, not run on hardware
//
#include <hip/hip_runtime.h>
#include <hip/hip_bf16.h>

typedef _Float16 f16;
typedef __attribute__((ext_vector_type(16))) _Float16 v16h;
typedef __attribute__((ext_vector_type(8)))  _Float16 v8h;
typedef __attribute__((ext_vector_type(8)))  float    v8f;

#define N_TAGS  1024
#define NWG     32            // workgroups; each owns WGROWS rows of E, LDS-resident
#define WGROWS  32            // NWG * WGROWS == N_TAGS
#define NTH     256           // 8 wave32s
#define ESTRIDE (N_TAGS + 8)  // padded f16 row stride: 2064 B = 516 dw = 4 banks skew
                              //  -> 16 consecutive rows * b128 cover all 64 banks once

// ------------- prep: E = exp(transitions) split into f16 hi/lo; init state -------------
__global__ void crf_prep(const float* __restrict__ trans,
                         const int* __restrict__ start_idx,
                         f16* __restrict__ Eh, f16* __restrict__ El,
                         float* __restrict__ aG, int* __restrict__ flags) {
  int i = blockIdx.x * blockDim.x + threadIdx.x;
  if (i < N_TAGS * N_TAGS) {
    float e = __expf(trans[i]);
    f16  hi = (f16)e;
    Eh[i] = hi;
    El[i] = (f16)(e - (float)hi);
  }
  if (i < N_TAGS) {
    aG[i]          = (i == *start_idx) ? 0.0f : -1000.0f;  // parity-0 alphas
    aG[N_TAGS + i] = 0.0f;
  }
  if (i < NWG) flags[i] = 0;
}

// ------------------------------- sequential CRF scan ----------------------------------
__launch_bounds__(NTH, 1)
__global__ void crf_scan(const float* __restrict__ unary,
                         const float* __restrict__ trans,
                         const int* __restrict__ end_idx_p,
                         const f16* __restrict__ Eh,
                         const f16* __restrict__ El,
                         float* __restrict__ aG,     // [2][N_TAGS] double-buffered alphas
                         int*   __restrict__ flags,  // [NWG] steps completed per WG
                         float* __restrict__ out,
                         int T) {
  extern __shared__ __align__(32) char smem[];
  f16*   sEh  = (f16*)smem;                     // [WGROWS][ESTRIDE] resident E_hi rows
  f16*   sEl  = sEh + WGROWS * ESTRIDE;         // [WGROWS][ESTRIDE] resident E_lo rows
  f16*   sPh  = sEl + WGROWS * ESTRIDE;         // [N_TAGS] p_hi
  f16*   sPl  = sPh + N_TAGS;                   // [N_TAGS] p_lo
  float* sOut = (float*)(sPl + N_TAGS);         // [WGROWS] row sums
  float* sRed = sOut + WGROWS;                  // [8] per-wave reduction slots
  float* sM   = sRed + 8;                       // [1] broadcast max

  const int tid  = threadIdx.x;
  const int wave = tid >> 5;
  const int lane = tid & 31;
  const int wg   = blockIdx.x;
  const int r0   = wg * WGROWS;

  // one-time: park this WG's slice of E in LDS (padded rows) for the whole scan
  {
    const int per_row = N_TAGS / 8;             // uint4 (= 8 f16) per row
    for (int k = tid; k < WGROWS * per_row; k += NTH) {
      const int row = k / per_row, col = k % per_row;
      ((uint4*)(sEh + row * ESTRIDE))[col] =
          ((const uint4*)(Eh + (size_t)(r0 + row) * N_TAGS))[col];
      ((uint4*)(sEl + row * ESTRIDE))[col] =
          ((const uint4*)(El + (size_t)(r0 + row) * N_TAGS))[col];
    }
  }
  __syncthreads();

  const int j0 = tid * (N_TAGS / NTH);   // 4 alpha entries per thread
  double Cd = 0.0;                       // running log-normalizer (f64: ~5e5 at T)

  // branchless symmetric work split: 8 waves = 2 tiles x 4 products
  const int tile = wave & 1;                    // 16-row tile of this WG
  const int prod = wave >> 1;                   // 0:Eh*ph 1:Eh*pl 2:El*ph 3:El*pl
  const int half = lane >> 4;
  const f16* Erow  = ((prod >> 1) ? sEl : sEh) + (tile * 16 + (lane & 15)) * ESTRIDE;
  const f16* Pbase = (prod & 1) ? sPl : sPh;

  for (int t = 0; t < T; ++t) {
    const int par = t & 1;

    // wait until every producer WG published alphas for step t
    if (wave == 0 && lane < NWG) {
      while (__hip_atomic_load(&flags[lane], __ATOMIC_ACQUIRE,
                               __HIP_MEMORY_SCOPE_AGENT) < t)
        __builtin_amdgcn_s_sleep(1);
    }
    __syncthreads();

    // fetch alphas from L2 (agent scope -> fresh)
    float av[4];
    #pragma unroll
    for (int k = 0; k < 4; ++k)
      av[k] = __hip_atomic_load(&aG[par * N_TAGS + j0 + k], __ATOMIC_RELAXED,
                                __HIP_MEMORY_SCOPE_AGENT);

    // m = max_j alpha[j]  (shfl within wave32, then cross-wave via LDS)
    float lm = fmaxf(fmaxf(av[0], av[1]), fmaxf(av[2], av[3]));
    #pragma unroll
    for (int o = 16; o > 0; o >>= 1) lm = fmaxf(lm, __shfl_xor(lm, o, 32));
    if (lane == 0) sRed[wave] = lm;
    if (tid < WGROWS) sOut[tid] = 0.0f;
    __syncthreads();
    if (tid == 0) {
      float m = sRed[0];
      #pragma unroll
      for (int i = 1; i < NTH / 32; ++i) m = fmaxf(m, sRed[i]);
      *sM = m;
    }
    __syncthreads();
    const float m = *sM;
    Cd += (double)m;

    // p = exp(alpha - m) split into f16 hi/lo
    #pragma unroll
    for (int k = 0; k < 4; ++k) {
      float p = __expf(av[k] - m);
      f16 hi = (f16)p;
      sPh[j0 + k] = hi;
      sPl[j0 + k] = (f16)(p - (float)hi);
    }
    __syncthreads();

    // s = E*p: each wave runs one of the four split products over full K
    // via v_wmma_f32_16x16x32_f16 (p broadcast across the 16 B columns)
    {
      v8f acc = {};
      #pragma unroll 4
      for (int c = 0; c < N_TAGS / 32; ++c) {
        const int kbase = c * 32;
        // A 16x32 f16: lane<16 holds K {0..7,16..23}, lane>=16 holds {8..15,24..31}
        v8h alo = *(const v8h*)(Erow + kbase + half * 8);
        v8h ahi = *(const v8h*)(Erow + kbase + half * 8 + 16);
        v16h a;
        #pragma unroll
        for (int i = 0; i < 8; ++i) { a[i] = alo[i]; a[i + 8] = ahi[i]; }
        // B 32x16 f16: lane<16 holds K {0..7,16..23}, lane>=16 holds {8..15,24..31}
        v8h blo = *(const v8h*)(Pbase + kbase + half * 8);
        v8h bhi = *(const v8h*)(Pbase + kbase + half * 8 + 16);
        v16h b;
        #pragma unroll
        for (int i = 0; i < 8; ++i) { b[i] = blo[i]; b[i + 8] = bhi[i]; }
        acc = __builtin_amdgcn_wmma_f32_16x16x32_f16(false, a, false, b,
                                                     (short)0, acc, false, false);
      }
      // D layout: VGPR r -> row r (lanes 0..15) / row r+8 (lanes 16..31); cols equal
      if ((lane & 15) == 0) {
        const int rbase = tile * 16 + half * 8;
        #pragma unroll
        for (int r = 0; r < 8; ++r) atomicAdd(&sOut[rbase + r], acc[r]);  // ds_add_f32
      }
    }
    __syncthreads();

    // alpha'[i] = log(s_i) + unary_t[i]; publish chunk + prefetch next unary row
    if (tid < WGROWS) {
      const int grow = r0 + tid;
      float an = __logf(sOut[tid]) + unary[(size_t)t * N_TAGS + grow];
      __hip_atomic_store(&aG[((t + 1) & 1) * N_TAGS + grow], an,
                         __ATOMIC_RELAXED, __HIP_MEMORY_SCOPE_AGENT);
      if (t + 1 < T)
        __builtin_prefetch(&unary[(size_t)(t + 1) * N_TAGS + grow], 0, 0);
    }
    __threadfence();
    __syncthreads();
    if (tid == 0)
      __hip_atomic_store(&flags[wg], t + 1, __ATOMIC_RELEASE,
                         __HIP_MEMORY_SCOPE_AGENT);
  }

  // ---- terminal logsumexp on WG 0: out = C + LSE_i(a_T[i] + trans[end, i]) ----
  if (wg != 0) return;
  if (wave == 0 && lane < NWG) {
    while (__hip_atomic_load(&flags[lane], __ATOMIC_ACQUIRE,
                             __HIP_MEMORY_SCOPE_AGENT) < T)
      __builtin_amdgcn_s_sleep(1);
  }
  __syncthreads();
  const int parF = T & 1;
  const int eidx = *end_idx_p;
  float tv[4];
  #pragma unroll
  for (int k = 0; k < 4; ++k)
    tv[k] = __hip_atomic_load(&aG[parF * N_TAGS + j0 + k], __ATOMIC_RELAXED,
                              __HIP_MEMORY_SCOPE_AGENT)
          + trans[(size_t)eidx * N_TAGS + j0 + k];

  float lm2 = fmaxf(fmaxf(tv[0], tv[1]), fmaxf(tv[2], tv[3]));
  #pragma unroll
  for (int o = 16; o > 0; o >>= 1) lm2 = fmaxf(lm2, __shfl_xor(lm2, o, 32));
  if (lane == 0) sRed[wave] = lm2;
  __syncthreads();
  float m2 = sRed[0];
  #pragma unroll
  for (int i = 1; i < NTH / 32; ++i) m2 = fmaxf(m2, sRed[i]);

  float ls = 0.0f;
  #pragma unroll
  for (int k = 0; k < 4; ++k) ls += __expf(tv[k] - m2);
  #pragma unroll
  for (int o = 16; o > 0; o >>= 1) ls += __shfl_xor(ls, o, 32);
  if (lane == 0) sOut[wave] = ls;
  __syncthreads();
  if (tid == 0) {
    float S = 0.0f;
    #pragma unroll
    for (int i = 0; i < NTH / 32; ++i) S += sOut[i];
    out[0] = (float)(Cd + (double)m2 + (double)__logf(S));
  }
}

extern "C" void kernel_launch(void* const* d_in, const int* in_sizes, int n_in,
                              void* d_out, int out_size, void* d_ws, size_t ws_size,
                              hipStream_t stream) {
  const float* unary     = (const float*)d_in[0];
  const float* trans     = (const float*)d_in[1];
  const int*   start_idx = (const int*)d_in[2];
  const int*   end_idx   = (const int*)d_in[3];
  float* out = (float*)d_out;
  const int T = in_sizes[0] / N_TAGS;

  // workspace: Eh (2MB) | El (2MB) | aG (8KB) | flags
  char* ws = (char*)d_ws;
  f16*   Eh    = (f16*)ws;
  f16*   El    = (f16*)(ws + (size_t)N_TAGS * N_TAGS * sizeof(f16));
  float* aG    = (float*)(ws + (size_t)N_TAGS * N_TAGS * 2 * sizeof(f16));
  int*   flags = (int*)(ws + (size_t)N_TAGS * N_TAGS * 2 * sizeof(f16)
                           + (size_t)2 * N_TAGS * sizeof(float));

  crf_prep<<<(N_TAGS * N_TAGS + 255) / 256, 256, 0, stream>>>(
      trans, start_idx, Eh, El, aG, flags);

  const size_t shbytes =
      (size_t)(2 * WGROWS * ESTRIDE + 2 * N_TAGS) * sizeof(f16) +
      (size_t)(WGROWS + 8 + 8) * sizeof(float);
  crf_scan<<<NWG, NTH, shbytes, stream>>>(
      unary, trans, end_idx, Eh, El, aG, flags, out, T);
}
